// ASTGCN_block_1632087572509
// MI455X (gfx1250) — matrix operand order repaired and checked by reading, not compiled, not run on hardware
//
#include <hip/hip_runtime.h>
#include <hip/hip_bf16.h>
#include <stdint.h>

#define B_   16
#define N_   1024
#define FIN_ 32
#define T_   24
#define K_   3
#define CC_  64
#define CT_  64
#define FT_  768           // FIN_*T_
#define EPS_ 1e-5f

typedef __attribute__((ext_vector_type(16))) __bf16 v16bf;
typedef __attribute__((ext_vector_type(8)))  float  v8f;

// ---------------------------------------------------------------------------
// 1. tmp[b,t,f] = sum_n x[b,n,f,t] * U1[n]
__global__ void k_tmp(const float* __restrict__ x, const float* __restrict__ U1,
                      float* __restrict__ tmp) {
    int idx = blockIdx.x * blockDim.x + threadIdx.x;       // B*T*F = 12288
    if (idx >= B_ * T_ * FIN_) return;
    int b = idx / (T_ * FIN_);
    int r = idx % (T_ * FIN_);
    int t = r / FIN_, f = r % FIN_;
    float acc = 0.f;
    for (int n = 0; n < N_; ++n)
        acc += x[(((long)b * N_ + n) * FIN_ + f) * T_ + t] * U1[n];
    tmp[idx] = acc;
}

// 2. lhs2[b,t,m] = sum_f tmp[b,t,f] * U2[f,m]
__global__ void k_lhs2(const float* __restrict__ tmp, const float* __restrict__ U2,
                       float* __restrict__ lhs2) {
    int idx = blockIdx.x * blockDim.x + threadIdx.x;       // B*T*N
    if (idx >= B_ * T_ * N_) return;
    int b = idx / (T_ * N_);
    int r = idx % (T_ * N_);
    int t = r / N_, m = r % N_;
    float acc = 0.f;
#pragma unroll
    for (int f = 0; f < FIN_; ++f)
        acc += tmp[(b * T_ + t) * FIN_ + f] * U2[f * N_ + m];
    lhs2[idx] = acc;
}

// 3. rhs_t[b,n,t] = sum_f U3[f] * x[b,n,f,t]
__global__ void k_rhst(const float* __restrict__ x, const float* __restrict__ U3,
                       float* __restrict__ rhs_t) {
    int idx = blockIdx.x * blockDim.x + threadIdx.x;       // B*N*T
    if (idx >= B_ * N_ * T_) return;
    int b = idx / (N_ * T_);
    int r = idx % (N_ * T_);
    int n = r / T_, t = r % T_;
    float acc = 0.f;
#pragma unroll
    for (int f = 0; f < FIN_; ++f)
        acc += U3[f] * x[(((long)b * N_ + n) * FIN_ + f) * T_ + t];
    rhs_t[idx] = acc;
}

// 4. E[b,t,s] = softmax_t( sum_n lhs2[b,t,n]*rhs_t[b,n,s] )   (one block per (s,b))
__global__ void k_E(const float* __restrict__ lhs2, const float* __restrict__ rhs_t,
                    float* __restrict__ E) {
    __shared__ float e_sh[T_];
    __shared__ float ex_sh[T_];
    int s = blockIdx.x, b = blockIdx.y;
    int t = threadIdx.x;                                   // 32 threads, 24 active
    if (t < T_) {
        float acc = 0.f;
        const float* lrow = lhs2 + (b * T_ + t) * N_;
        for (int n = 0; n < N_; ++n)
            acc += lrow[n] * rhs_t[((long)b * N_ + n) * T_ + s];
        e_sh[t] = acc;
    }
    __syncthreads();
    if (t < T_) {
        float mx = -1e30f;
        for (int i = 0; i < T_; ++i) mx = fmaxf(mx, e_sh[i]);
        ex_sh[t] = __expf(e_sh[t] - mx);
    }
    __syncthreads();
    if (t < T_) {
        float sm = 0.f;
        for (int i = 0; i < T_; ++i) sm += ex_sh[i];
        E[(b * T_ + t) * T_ + s] = ex_sh[t] / sm;
    }
}

// 5. x_TAt[b,n,f,s] = sum_t x[b,n,f,t]*E[b,t,s]   (one block per (n,b))
__global__ void k_xTAt(const float* __restrict__ x, const float* __restrict__ E,
                       float* __restrict__ xTAt) {
    __shared__ float xr[FT_];
    __shared__ float Es[T_ * T_];
    int n = blockIdx.x, b = blockIdx.y, tid = threadIdx.x;
    const long base = ((long)b * N_ + n) * FT_;
    for (int j = tid; j < FT_; j += 256) xr[j] = x[base + j];
    for (int j = tid; j < T_ * T_; j += 256) Es[j] = E[b * T_ * T_ + j];
    __syncthreads();
    for (int j = tid; j < FT_; j += 256) {
        int f = j / T_, s = j % T_;
        float acc = 0.f;
#pragma unroll
        for (int t = 0; t < T_; ++t) acc += xr[f * T_ + t] * Es[t * T_ + s];
        xTAt[base + j] = acc;
    }
}

// 6. lhs_s[b,n,t'] and rhs_sT[b,n,t]   (one block of 32 per (n,b))
__global__ void k_spat1(const float* __restrict__ xTAt, const float* __restrict__ W1,
                        const float* __restrict__ W2, const float* __restrict__ W3,
                        float* __restrict__ lhs_s, float* __restrict__ rhs_sT) {
    __shared__ float xr[FT_];
    __shared__ float t2[FIN_];
    int n = blockIdx.x, b = blockIdx.y, tid = threadIdx.x;
    const long base = ((long)b * N_ + n) * FT_;
    for (int j = tid; j < FT_; j += 32) xr[j] = xTAt[base + j];
    __syncthreads();
    {   // tmp2[f] = sum_t xr[f,t]*W1[t]
        float s = 0.f;
#pragma unroll
        for (int t = 0; t < T_; ++t) s += xr[tid * T_ + t] * W1[t];
        t2[tid] = s;
    }
    __syncthreads();
    if (tid < T_) {
        float a = 0.f, r = 0.f;
#pragma unroll
        for (int f = 0; f < FIN_; ++f) {
            a += t2[f] * W2[f * T_ + tid];
            r += W3[f] * xr[f * T_ + tid];
        }
        lhs_s[((long)b * N_ + n) * T_ + tid]  = a;
        rhs_sT[((long)b * N_ + n) * T_ + tid] = r;
    }
}

// 7. St[b,m,n] = softmax_n( sum_t lhs_s[b,n,t]*rhs_sT[b,m,t] )  (block per (m,b))
__global__ void k_spat2(const float* __restrict__ lhs_s, const float* __restrict__ rhs_sT,
                        float* __restrict__ St) {
    __shared__ float r_sh[T_];
    __shared__ float red[256];
    int m = blockIdx.x, b = blockIdx.y, tid = threadIdx.x;
    if (tid < T_) r_sh[tid] = rhs_sT[((long)b * N_ + m) * T_ + tid];
    __syncthreads();
    float v[4];
    float mx = -1e30f;
#pragma unroll
    for (int j = 0; j < 4; ++j) {
        int n = tid + j * 256;
        float s = 0.f;
        const float* lrow = lhs_s + ((long)b * N_ + n) * T_;
#pragma unroll
        for (int t = 0; t < T_; ++t) s += lrow[t] * r_sh[t];
        v[j] = s;
        mx = fmaxf(mx, s);
    }
    red[tid] = mx;
    __syncthreads();
    for (int s = 128; s > 0; s >>= 1) {
        if (tid < s) red[tid] = fmaxf(red[tid], red[tid + s]);
        __syncthreads();
    }
    mx = red[0];
    __syncthreads();
    float ssum = 0.f;
#pragma unroll
    for (int j = 0; j < 4; ++j) { v[j] = __expf(v[j] - mx); ssum += v[j]; }
    red[tid] = ssum;
    __syncthreads();
    for (int s = 128; s > 0; s >>= 1) {
        if (tid < s) red[tid] += red[tid + s];
        __syncthreads();
    }
    float inv = 1.f / red[0];
#pragma unroll
    for (int j = 0; j < 4; ++j)
        St[((long)b * N_ + m) * N_ + tid + j * 256] = v[j] * inv;
}

// 8a. chebT[k,m,n] = cheb[k,n,m]
__global__ void k_chebT(const float* __restrict__ cheb, float* __restrict__ chebT) {
    long idx = (long)blockIdx.x * blockDim.x + threadIdx.x;   // K*N*N
    if (idx >= (long)K_ * N_ * N_) return;
    long k = idx / ((long)N_ * N_);
    long r = idx % ((long)N_ * N_);
    long m = r / N_, n = r % N_;
    chebT[idx] = cheb[(k * N_ + n) * N_ + m];
}

// 8b. xbfT[b,c,n] = bf16( x[b,n,c] )   (column-major bf16 copy of x, n contiguous)
__global__ void k_xbfT(const float* __restrict__ x, __bf16* __restrict__ xbfT) {
    long idx = (long)blockIdx.x * blockDim.x + threadIdx.x;   // B*FT*N (dest index)
    if (idx >= (long)B_ * FT_ * N_) return;
    long b = idx / ((long)FT_ * N_);
    long r = idx % ((long)FT_ * N_);
    long c = r / N_, n = r % N_;
    xbfT[idx] = (__bf16)x[((long)b * N_ + n) * FT_ + c];
}

// ---------------------------------------------------------------------------
// 9. Fused Chebyshev graph conv (the heavy GEMM) — WMMA bf16, fp32 accumulate.
//    One block per (16-row m-tile, b). P_k = (chebT_k ⊙ St_b)[16,1024] @ X_b[1024,768]
//    then out[16,64,24] += P_k ×_f Theta_k, ReLU, store sg[b,n,o,t].
//
//    A panel (16x1024) is staged per-k in LDS, pre-swizzled so each lane's
//    16-element fragment is one contiguous 32B LDS read. B fragments are
//    contiguous 32B runs in the transposed global bf16 copy of x.
//    The K-loop is kept rolled (#pragma unroll 1) so live fragments stay in
//    VGPRs — full unrolling previously spilled all B fragments to scratch.
__global__ void k_gcn_wmma(const float* __restrict__ chebT, const float* __restrict__ St,
                           const __bf16* __restrict__ xbfT, const float* __restrict__ Theta,
                           float* __restrict__ sg) {
    extern __shared__ char smem[];
    float*  out_sh = (float*)smem;                          // 16*64*24 = 24576 f (96KB)
    float*  p_sh   = out_sh + 16 * CC_ * T_;                // 16*768   = 12288 f (48KB)
    float*  th_sh  = p_sh + 16 * FT_;                       // 32*64    =  2048 f ( 8KB)
    __bf16* a_sh   = (__bf16*)(th_sh + FIN_ * CC_);         // 16*1024 bf16      (32KB)

    const int tid  = threadIdx.x;
    const int lane = tid & 31;
    const int wv   = tid >> 5;          // 8 waves, each owns 96 of the 768 cols
    const int half = lane >> 4;
    const int l16  = lane & 15;
    const int m0   = blockIdx.x * 16;
    const int b    = blockIdx.y;

    for (int j = tid; j < 16 * CC_ * T_; j += 256) out_sh[j] = 0.f;

    for (int k = 0; k < K_; ++k) {
        __syncthreads();
        // ---- stage full A panel for this k, swizzled for contiguous fragments.
        // element (m, n): K-in-chunk kk = n&31; slot = kk (contiguous order,
        // K = 16*half + e).  Lane (m,half) then reads 16
        // contiguous bf16 at chunk*512 + m*32 + half*16.
        for (int j = tid; j < 16 * N_; j += 256) {
            int m = j >> 10, n = j & (N_ - 1);
            int kk = n & 31;
            int slot = kk;
            float av = chebT[((long)k * N_ + (m0 + m)) * N_ + n] *
                       St[((long)b * N_ + (m0 + m)) * N_ + n];
            a_sh[(n >> 5) * 512 + m * 32 + slot] = (__bf16)av;
        }
        __syncthreads();

        v8f acc[6];
#pragma unroll
        for (int tt = 0; tt < 6; ++tt)
#pragma unroll
            for (int r = 0; r < 8; ++r) acc[tt][r] = 0.f;

        const __bf16* bcol = xbfT + ((long)b * FT_ + wv * 96 + l16) * N_ + 16 * half;
#pragma unroll 1
        for (int n0 = 0; n0 < N_; n0 += 32) {
            // A fragment: one contiguous 32B LDS read per lane
            v16bf afrag = *(const v16bf*)(a_sh + (n0 >> 5) * 512 + l16 * 32 + half * 16);
            // prefetch next chunk's B columns (global_prefetch_b8)
            if (n0 + 32 < N_) {
#pragma unroll
                for (int tt = 0; tt < 6; ++tt)
                    __builtin_prefetch(bcol + (long)tt * 16 * N_ + n0 + 32, 0, 0);
            }
#pragma unroll
            for (int tt = 0; tt < 6; ++tt) {
                // B fragment: K = e + 16*half -> 16 consecutive n of column col
                v16bf bfrag = *(const v16bf*)(bcol + (long)tt * 16 * N_ + n0);
                acc[tt] = __builtin_amdgcn_wmma_f32_16x16x32_bf16(
                    false, afrag, false, bfrag, (short)0, acc[tt], false, false);
            }
        }
        __syncthreads();
        // dump P_k to LDS: C layout VGPR r -> M = r + 8*half, N = l&15
#pragma unroll
        for (int tt = 0; tt < 6; ++tt) {
            int col = wv * 96 + tt * 16 + l16;
#pragma unroll
            for (int r = 0; r < 8; ++r)
                p_sh[(r + 8 * half) * FT_ + col] = acc[tt][r];
        }
        for (int j = tid; j < FIN_ * CC_; j += 256)
            th_sh[j] = Theta[k * FIN_ * CC_ + j];
        __syncthreads();
        // out[m,o,t] += sum_f P_k[m,f,t] * Theta_k[f,o]
        for (int j = tid; j < 16 * CC_ * T_; j += 256) {
            int t = j % T_;
            int o = (j / T_) % CC_;
            int m = j / (T_ * CC_);
            float s = 0.f;
#pragma unroll
            for (int f = 0; f < FIN_; ++f)
                s += p_sh[m * FT_ + f * T_ + t] * th_sh[f * CC_ + o];
            out_sh[j] += s;
        }
    }
    __syncthreads();
    for (int j = tid; j < 16 * CC_ * T_; j += 256) {
        int t = j % T_;
        int o = (j / T_) % CC_;
        int m = j / (T_ * CC_);
        float v = out_sh[j];
        sg[(((long)b * N_ + (m0 + m)) * CC_ + o) * T_ + t] = v > 0.f ? v : 0.f;
    }
}

// ---------------------------------------------------------------------------
// 10. time conv (1,3) pad 1 + residual 1x1 + ReLU + LayerNorm over CT
//     one block of 64 threads (ct) per (n,b)
__global__ void k_final(const float* __restrict__ sg, const float* __restrict__ x,
                        const float* __restrict__ tc_w, const float* __restrict__ tc_b,
                        const float* __restrict__ rc_w, const float* __restrict__ rc_b,
                        const float* __restrict__ gamma, const float* __restrict__ beta,
                        float* __restrict__ out) {
    __shared__ float sgr[CC_ * T_];   // sg[b,n,:,:]
    __shared__ float xr[FT_];
    __shared__ float z_sh[CT_];
    int n = blockIdx.x, b = blockIdx.y, ct = threadIdx.x;
    const long base = ((long)b * N_ + n);
    for (int j = ct; j < CC_ * T_; j += CT_) sgr[j] = sg[base * CC_ * T_ + j];
    for (int j = ct; j < FT_; j += CT_) xr[j] = x[base * FT_ + j];
    __syncthreads();
    float tcb = tc_b[ct], rcb = rc_b[ct], g = gamma[ct], be = beta[ct];
    for (int t = 0; t < T_; ++t) {
        float tco = tcb;
        for (int cc = 0; cc < CC_; ++cc) {
            const float* wp = tc_w + (ct * CC_ + cc) * 3;
            float s0 = (t > 0)      ? sgr[cc * T_ + t - 1] : 0.f;
            float s1 =                sgr[cc * T_ + t];
            float s2 = (t < T_ - 1) ? sgr[cc * T_ + t + 1] : 0.f;
            tco += s0 * wp[0] + s1 * wp[1] + s2 * wp[2];
        }
        float res = rcb;
#pragma unroll
        for (int f = 0; f < FIN_; ++f) res += xr[f * T_ + t] * rc_w[ct * FIN_ + f];
        float z = res + tco;
        z = z > 0.f ? z : 0.f;
        z_sh[ct] = z;
        __syncthreads();
        float mu = 0.f;
        for (int i = 0; i < CT_; ++i) mu += z_sh[i];
        mu *= (1.f / CT_);
        float var = 0.f;
        for (int i = 0; i < CT_; ++i) { float d = z_sh[i] - mu; var += d * d; }
        var *= (1.f / CT_);
        out[(base * CT_ + ct) * T_ + t] = (z - mu) * rsqrtf(var + EPS_) * g + be;
        __syncthreads();
    }
}

// ---------------------------------------------------------------------------
extern "C" void kernel_launch(void* const* d_in, const int* in_sizes, int n_in,
                              void* d_out, int out_size, void* d_ws, size_t ws_size,
                              hipStream_t stream) {
    (void)in_sizes; (void)n_in; (void)out_size; (void)ws_size;
    const float* x     = (const float*)d_in[0];
    const float* W1    = (const float*)d_in[1];
    const float* W2    = (const float*)d_in[2];
    const float* W3    = (const float*)d_in[3];
    const float* U1    = (const float*)d_in[4];
    const float* U2    = (const float*)d_in[5];
    const float* U3    = (const float*)d_in[6];
    const float* cheb  = (const float*)d_in[7];
    const float* Theta = (const float*)d_in[8];
    const float* tc_w  = (const float*)d_in[9];
    const float* tc_b  = (const float*)d_in[10];
    const float* rc_w  = (const float*)d_in[11];
    const float* rc_b  = (const float*)d_in[12];
    const float* gamma = (const float*)d_in[13];
    const float* beta  = (const float*)d_in[14];
    float* out = (float*)d_out;

    char* w = (char*)d_ws;
    float* tmp    = (float*)w; w += (size_t)B_ * T_ * FIN_ * 4;        // 12288
    float* lhs2   = (float*)w; w += (size_t)B_ * T_ * N_ * 4;          // 393216
    float* rhs_t  = (float*)w; w += (size_t)B_ * N_ * T_ * 4;
    float* E      = (float*)w; w += (size_t)B_ * T_ * T_ * 4;
    float* xTAt   = (float*)w; w += (size_t)B_ * N_ * FT_ * 4;         // 50MB
    float* lhs_s  = (float*)w; w += (size_t)B_ * N_ * T_ * 4;
    float* rhs_sT = (float*)w; w += (size_t)B_ * N_ * T_ * 4;
    float* St     = (float*)w; w += (size_t)B_ * N_ * N_ * 4;          // 67MB
    float* chebT  = (float*)w; w += (size_t)K_ * N_ * N_ * 4;          // 12.6MB
    float* sg     = (float*)w; w += (size_t)B_ * N_ * CC_ * T_ * 4;    // 100MB
    __bf16* xbfT  = (__bf16*)w;                                        // 25MB

    // temporal attention
    k_tmp <<<(B_ * T_ * FIN_ + 255) / 256, 256, 0, stream>>>(x, U1, tmp);
    k_lhs2<<<(B_ * T_ * N_   + 255) / 256, 256, 0, stream>>>(tmp, U2, lhs2);
    k_rhst<<<(B_ * N_ * T_   + 255) / 256, 256, 0, stream>>>(x, U3, rhs_t);
    k_E   <<<dim3(T_, B_), 32, 0, stream>>>(lhs2, rhs_t, E);
    k_xTAt<<<dim3(N_, B_), 256, 0, stream>>>(x, E, xTAt);
    // spatial attention
    k_spat1<<<dim3(N_, B_), 32, 0, stream>>>(xTAt, W1, W2, W3, lhs_s, rhs_sT);
    k_spat2<<<dim3(N_, B_), 256, 0, stream>>>(lhs_s, rhs_sT, St);
    // layout prep
    k_chebT<<<(K_ * N_ * N_ + 255) / 256, 256, 0, stream>>>(cheb, chebT);
    k_xbfT <<<((long)B_ * N_ * FT_ + 255) / 256, 256, 0, stream>>>(x, xbfT);
    // fused K-order graph conv via WMMA bf16 (fp32 accumulate)
    size_t lds = (size_t)(16 * CC_ * T_ + 16 * FT_ + FIN_ * CC_) * 4
               + (size_t)(16 * N_) * 2;                                // ~184 KB
    k_gcn_wmma<<<dim3(N_ / 16, B_), 256, lds, stream>>>(chebT, St, xbfT, Theta, sg);
    // time conv + residual + relu + layernorm
    k_final<<<dim3(N_, B_), CT_, 0, stream>>>(sg, x, tc_w, tc_b, rc_w, rc_b,
                                              gamma, beta, out);
}
